// MyGRU_18700287607631
// MI455X (gfx1250) — matrix operand order repaired and checked by reading, not compiled, not run on hardware
//
#include <hip/hip_runtime.h>

// GRU: B=64, T=1024, INPUT=512, HIDDEN=1024 (fp32 in/out, bf16 WMMA compute)
#define BB 64
#define TT 1024
#define II 512
#define HH 1024
#define NBLK 64          // persistent recurrent kernel: 64 blocks x 256 threads

typedef __attribute__((ext_vector_type(16))) __bf16       v16bf;
typedef __attribute__((ext_vector_type(8)))  float        v8f;
typedef __attribute__((ext_vector_type(4)))  unsigned int u32x4;
typedef __attribute__((ext_vector_type(4)))  float        f32x4;

union V16U {
    u32x4  q[2];   // two b128 loads
    __bf16 b[16];  // element view
    v16bf  v;      // WMMA operand view
};

static __device__ __forceinline__ float sigmoidf_(float x) {
    return 1.0f / (1.0f + __expf(-x));
}
static __device__ __forceinline__ float tanhf_(float x) {
    return 2.0f / (1.0f + __expf(-2.0f * x)) - 1.0f;
}
static __device__ __forceinline__ v8f wmma_bf16(v16bf a, v16bf b, v8f c) {
    // v_wmma_f32_16x16x32_bf16 : D = A(16x32) x B(32x16) + C(16x16 f32)
    return __builtin_amdgcn_wmma_f32_16x16x32_bf16(false, a, false, b,
                                                   (short)0, c, false, false);
}

// Grid-wide barrier: monotonic counter, unique target per use (no reset race).
static __device__ __forceinline__ void gbar(unsigned int* cnt, unsigned int target) {
    __threadfence();          // release our stores to device scope
    __syncthreads();
    if (threadIdx.x == 0) {
        atomicAdd(cnt, 1u);
        while (__hip_atomic_load(cnt, __ATOMIC_RELAXED, __HIP_MEMORY_SCOPE_AGENT) < target) {
            __builtin_amdgcn_s_sleep(1);
        }
    }
    __syncthreads();
    __threadfence();          // acquire other blocks' stores
}

// ---------------------------------------------------------------------------
// Kernel 1: fp32 -> bf16 conversion of weights + h0, zero the barrier counter.
// ---------------------------------------------------------------------------
__global__ void cvt_kernel(const float* __restrict__ Wz, const float* __restrict__ Wr,
                           const float* __restrict__ Wh, const float* __restrict__ Uz,
                           const float* __restrict__ Ur, const float* __restrict__ Uh,
                           const float* __restrict__ h0,
                           __bf16* __restrict__ Wzb, __bf16* __restrict__ Wrb,
                           __bf16* __restrict__ Whb, __bf16* __restrict__ Uzb,
                           __bf16* __restrict__ Urb, __bf16* __restrict__ Uhb,
                           __bf16* __restrict__ hbuf, unsigned int* __restrict__ bar)
{
    long i = (long)blockIdx.x * blockDim.x + threadIdx.x;
    if (i == 0) *bar = 0u;                          // re-zero every launch (graph replay!)
    const long NW = (long)HH * II;                  // 524288
    const long NU = (long)HH * HH;                  // 1048576
    const long NH = (long)BB * HH;                  // 65536
    const long total  = 3 * NW + 3 * NU + NH;
    const long stride = (long)gridDim.x * blockDim.x;
    for (; i < total; i += stride) {
        long j = i;
        if (j < NW)              { Wzb[j]  = (__bf16)Wz[j]; }
        else if ((j -= NW) < NW) { Wrb[j]  = (__bf16)Wr[j]; }
        else if ((j -= NW) < NW) { Whb[j]  = (__bf16)Wh[j]; }
        else if ((j -= NW) < NU) { Uzb[j]  = (__bf16)Uz[j]; }
        else if ((j -= NU) < NU) { Urb[j]  = (__bf16)Ur[j]; }
        else if ((j -= NU) < NU) { Uhb[j]  = (__bf16)Uh[j]; }
        else { j -= NU;            hbuf[j] = (__bf16)h0[j]; }
    }
}

// ---------------------------------------------------------------------------
// Kernel 2: input projections  p[g][bt,h] = x[bt,:] . W_g[h,:] + b_g[h]
// Wave = one 16(M) x 64(N) strip (4 WMMA tiles, A reused in registers).
// Block = 8 waves = 32(M) x 256(N). grid = (BT/32, H/256, 3 gates).
// x (128MB) stays L2-resident; proj outputs are streaming -> NT stores.
// ---------------------------------------------------------------------------
__global__ void __launch_bounds__(256) proj_kernel(
    const float* __restrict__ x,
    const __bf16* __restrict__ Wzb, const __bf16* __restrict__ Wrb,
    const __bf16* __restrict__ Whb,
    const float* __restrict__ bz, const float* __restrict__ br, const float* __restrict__ bh,
    float* __restrict__ pz, float* __restrict__ pr, float* __restrict__ ph)
{
    const int lane = threadIdx.x & 31;
    const int wave = threadIdx.x >> 5;
    const int half = lane >> 4;
    const int l15  = lane & 15;
    const int g    = blockIdx.z;
    const __bf16* W    = (g == 0) ? Wzb : (g == 1) ? Wrb : Whb;
    const float*  bias = (g == 0) ? bz  : (g == 1) ? br  : bh;
    float*        out  = (g == 0) ? pz  : (g == 1) ? pr  : ph;

    const int base_m  = blockIdx.x * 32 + (wave >> 2) * 16;
    const int base_n0 = blockIdx.y * 256 + (wave & 3) * 64;

    const float* xrow = x + (long)(base_m + l15) * II;    // A row = bt index

    const __bf16* wrow[4];
    int ncol[4];
#pragma unroll
    for (int j = 0; j < 4; ++j) {
        ncol[j] = base_n0 + 16 * j + l15;                 // B column (hidden unit)
        wrow[j] = W + (long)ncol[j] * II + 8 * half;      // B[k,n]=W[n,k], chunks K=[8h,+8),[16+8h,+8)
    }

    v8f acc[4] = {};
    for (int k0 = 0; k0 < II; k0 += 32) {
        // A (16x32 bf16): lane chunks K=[k0+8h,+8) and K=[k0+16+8h,+8), cvt f32->bf16
        const float* ap = xrow + k0 + 8 * half;
        f32x4 f0 = *(const f32x4*)(ap);
        f32x4 f1 = *(const f32x4*)(ap + 4);
        f32x4 f2 = *(const f32x4*)(ap + 16);
        f32x4 f3 = *(const f32x4*)(ap + 20);
        V16U a;
#pragma unroll
        for (int e = 0; e < 4; ++e) {
            a.b[e]      = (__bf16)f0[e];
            a.b[4 + e]  = (__bf16)f1[e];
            a.b[8 + e]  = (__bf16)f2[e];
            a.b[12 + e] = (__bf16)f3[e];
        }
#pragma unroll
        for (int j = 0; j < 4; ++j) {
            V16U b;
            b.q[0] = *(const u32x4*)(wrow[j] + k0);       // K=[k0+8h,+8) and K=[k0+16+8h,+8)
            b.q[1] = *(const u32x4*)(wrow[j] + k0 + 16);
            acc[j] = wmma_bf16(a.v, b.v, acc[j]);
        }
    }

    const int rowb = base_m + 8 * half;                   // C element v -> row rowb+v
#pragma unroll
    for (int j = 0; j < 4; ++j) {
        float bn = bias[ncol[j]];
#pragma unroll
        for (int v = 0; v < 8; ++v) {
            __builtin_nontemporal_store(acc[j][v] + bn,
                                        &out[(long)(rowb + v) * HH + ncol[j]]);
        }
    }
}

// ---------------------------------------------------------------------------
// Kernel 3: persistent recurrent scan. 64 blocks x 8 waves = 512 waves.
// gate = gw&1 (0:z, 1:r); tile id = gw>>1 -> 4(M) x 64(N) 16x16 tiles.
// Phase A: z/r GEMMs (L2-resident h + U); write zbuf / rh bf16. Grid barrier.
// Phase B: h-tilde GEMM on rh, tanh, blend, store h (bf16) + out (f32). Barrier.
// Streaming pre-activation loads are hoisted ABOVE the K-loops so their HBM
// latency hides under the GEMM instead of sitting on the serial critical path.
// ---------------------------------------------------------------------------
__global__ void __launch_bounds__(256) gru_kernel(
    const __bf16* __restrict__ Uzb, const __bf16* __restrict__ Urb,
    const __bf16* __restrict__ Uhb,
    const float* __restrict__ pz, const float* __restrict__ pr, const float* __restrict__ ph,
    __bf16* __restrict__ hbuf, __bf16* __restrict__ rhbuf,
    float* __restrict__ zbuf,
    float* __restrict__ out, float* __restrict__ hn,
    unsigned int* __restrict__ bar)
{
    const int lane = threadIdx.x & 31;
    const int wave = threadIdx.x >> 5;
    const int half = lane >> 4;
    const int l15  = lane & 15;
    const int gw   = blockIdx.x * 8 + wave;     // 0..511
    const int gate = gw & 1;                    // 0 -> z (and h-tilde), 1 -> r
    const int id   = gw >> 1;                   // 0..255
    const int base_m = (id & 3) * 16;           // batch-row tile
    const int base_n = (id >> 2) * 16;          // hidden-col tile

    const int arow = base_m + l15;
    const int ncol = base_n + l15;
    const int rowb = base_m + 8 * half;

    const __bf16* Ug = gate ? Urb : Uzb;
    const float*  pg = gate ? pr  : pz;

    const __bf16* aH0  = hbuf  + (long)arow * HH + 8 * half;   // A from h
    const __bf16* aRH0 = rhbuf + (long)arow * HH + 8 * half;   // A from r*h
    const __bf16* bG   = Ug  + (long)ncol * HH + 8 * half;     // B = U_g^T col
    const __bf16* bH   = Uhb + (long)ncol * HH + 8 * half;

    const float* pgp = pg + (long)rowb * (TT * HH) + ncol;     // + t*HH + v*TT*HH
    const float* php = ph + (long)rowb * (TT * HH) + ncol;
    float*       outp = out + (long)rowb * (TT * HH) + ncol;

    unsigned int barIdx = 0;

    for (int t = 0; t < TT; ++t) {
        const long toff = (long)t * HH;

        // ---- phase A: pre-activation GEMM for z or r -----------------------
        // Hoisted: streaming gate bias (HBM, NT) + current h values (L2).
        float pre[8];
#pragma unroll
        for (int v = 0; v < 8; ++v)
            pre[v] = __builtin_nontemporal_load(pgp + toff + (long)v * (TT * HH));
        float hv8[8];
        if (gate) {
#pragma unroll
            for (int v = 0; v < 8; ++v)
                hv8[v] = (float)hbuf[(rowb + v) * HH + ncol];
        }

        v8f acc = {};
#pragma unroll 4
        for (int k0 = 0; k0 < HH; k0 += 32) {
            V16U a, b;
            a.q[0] = *(const u32x4*)(aH0 + k0);
            a.q[1] = *(const u32x4*)(aH0 + k0 + 16);
            b.q[0] = *(const u32x4*)(bG + k0);
            b.q[1] = *(const u32x4*)(bG + k0 + 16);
            acc = wmma_bf16(a.v, b.v, acc);
        }
        if (gate == 0) {
#pragma unroll
            for (int v = 0; v < 8; ++v) {
                float zv = sigmoidf_(acc[v] + pre[v]);
                zbuf[(rowb + v) * HH + ncol] = zv;
            }
        } else {
#pragma unroll
            for (int v = 0; v < 8; ++v) {
                float rv = sigmoidf_(acc[v] + pre[v]);
                rhbuf[(rowb + v) * HH + ncol] = (__bf16)(rv * hv8[v]);
            }
        }
        ++barIdx;
        gbar(bar, barIdx * (unsigned)NBLK);

        // ---- phase B: h-tilde GEMM + state update (gate-0 waves) -----------
        if (gate == 0) {
            float preh[8], zv8[8], hb8[8];
#pragma unroll
            for (int v = 0; v < 8; ++v) {
                preh[v] = __builtin_nontemporal_load(php + toff + (long)v * (TT * HH));
                zv8[v]  = zbuf[(rowb + v) * HH + ncol];
                hb8[v]  = (float)hbuf[(rowb + v) * HH + ncol];
            }

            v8f acch = {};
#pragma unroll 4
            for (int k0 = 0; k0 < HH; k0 += 32) {
                V16U a, b;
                a.q[0] = *(const u32x4*)(aRH0 + k0);
                a.q[1] = *(const u32x4*)(aRH0 + k0 + 16);
                b.q[0] = *(const u32x4*)(bH + k0);
                b.q[1] = *(const u32x4*)(bH + k0 + 16);
                acch = wmma_bf16(a.v, b.v, acch);
            }
#pragma unroll
            for (int v = 0; v < 8; ++v) {
                float ht   = tanhf_(acch[v] + preh[v]);
                float hnew = (1.0f - zv8[v]) * hb8[v] + zv8[v] * ht;
                hbuf[(rowb + v) * HH + ncol] = (__bf16)hnew;       // next-step A (L2)
                __builtin_nontemporal_store(hnew, (float*)(outp + toff + (long)v * (TT * HH)));
                if (t == TT - 1) hn[(rowb + v) * HH + ncol] = hnew;
            }
        }
        ++barIdx;
        gbar(bar, barIdx * (unsigned)NBLK);
    }
}

// ---------------------------------------------------------------------------
extern "C" void kernel_launch(void* const* d_in, const int* in_sizes, int n_in,
                              void* d_out, int out_size, void* d_ws, size_t ws_size,
                              hipStream_t stream)
{
    (void)in_sizes; (void)n_in; (void)out_size; (void)ws_size;

    const float* x  = (const float*)d_in[0];
    const float* h0 = (const float*)d_in[1];
    const float* Wz = (const float*)d_in[2];
    const float* bz = (const float*)d_in[3];
    const float* Uz = (const float*)d_in[4];
    const float* Wr = (const float*)d_in[5];
    const float* br = (const float*)d_in[6];
    const float* Ur = (const float*)d_in[7];
    const float* Wh = (const float*)d_in[8];
    const float* bh = (const float*)d_in[9];
    const float* Uh = (const float*)d_in[10];

    float* out = (float*)d_out;
    float* hn  = out + (size_t)BB * TT * HH;

    char* ws = (char*)d_ws;
    size_t off = 0;
    auto take = [&](size_t bytes) -> char* {
        char* p = ws + off;
        off += (bytes + 255) & ~(size_t)255;
        return p;
    };

    const size_t PROJ = (size_t)BB * TT * HH;      // 67,108,864 elements
    float* pz = (float*)take(PROJ * 4);
    float* pr = (float*)take(PROJ * 4);
    float* ph = (float*)take(PROJ * 4);
    __bf16* Wzb = (__bf16*)take((size_t)HH * II * 2);
    __bf16* Wrb = (__bf16*)take((size_t)HH * II * 2);
    __bf16* Whb = (__bf16*)take((size_t)HH * II * 2);
    __bf16* Uzb = (__bf16*)take((size_t)HH * HH * 2);
    __bf16* Urb = (__bf16*)take((size_t)HH * HH * 2);
    __bf16* Uhb = (__bf16*)take((size_t)HH * HH * 2);
    __bf16* hbuf  = (__bf16*)take((size_t)BB * HH * 2);
    __bf16* rhbuf = (__bf16*)take((size_t)BB * HH * 2);
    float*  zbuf  = (float*)take((size_t)BB * HH * 4);
    unsigned int* bar = (unsigned int*)take(256);

    cvt_kernel<<<4096, 256, 0, stream>>>(Wz, Wr, Wh, Uz, Ur, Uh, h0,
                                         Wzb, Wrb, Whb, Uzb, Urb, Uhb, hbuf, bar);

    dim3 pgrid((BB * TT) / 32, HH / 256, 3);       // (2048, 4, 3)
    proj_kernel<<<pgrid, 256, 0, stream>>>(x, Wzb, Wrb, Whb, bz, br, bh, pz, pr, ph);

    gru_kernel<<<NBLK, 256, 0, stream>>>(Uzb, Urb, Uhb, pz, pr, ph,
                                         hbuf, rhbuf, zbuf, out, hn, bar);
}
